// CharRNN_27779848471276
// MI455X (gfx1250) — matrix operand order repaired and checked by reading, not compiled, not run on hardware
//
#include <hip/hip_runtime.h>
#include <hip/hip_bf16.h>
#include <math.h>

// ---------------------------------------------------------------------------
// CharRNN on gfx1250: weights-stationary-in-LDS persistent RNN with bf16 WMMA.
//   step:  h = tanh([x;h] @ [W_ih;W_hh]^T)   (M=64, N=1024, K=1280)
//   final: out = h @ W_proj^T + b_proj       (64 x 128)
// 32 persistent workgroups x 256 threads (8 wave32s). Each WG owns 32 output
// columns; its 1280x32 bf16 weight slice lives in LDS for all 512 steps.
// Per-step h exchange uses GLOBAL_LOAD_ASYNC_TO_LDS (ASYNCcnt) when available.
// ---------------------------------------------------------------------------

typedef __attribute__((ext_vector_type(16))) __bf16 bf16x16;
typedef __attribute__((ext_vector_type(8)))  __bf16 bf16x8;
typedef __attribute__((ext_vector_type(4)))  __bf16 bf16x4;
typedef __attribute__((ext_vector_type(8)))  float  f32x8;
typedef int v4i __attribute__((vector_size(16)));   // payload type of *_b128

#define BATCH   64
#define SEQ     512
#define EMBED   256
#define HIDDEN  1024
#define NCHAR   128
#define KTOT    (EMBED + HIDDEN)      // 1280
#define NWG     32
#define COLS_WG (HIDDEN / NWG)        // 32 output columns per workgroup
#define THREADS 256

#define ASG __attribute__((address_space(1)))
#define ASL __attribute__((address_space(3)))

__device__ __forceinline__ float fast_tanh(float x) {
#if __has_builtin(__builtin_amdgcn_tanhf)
    return __builtin_amdgcn_tanhf(x);     // native V_TANH_F32 on gfx1250
#else
    return tanhf(x);
#endif
}

// workspace layout: [0..255] barrier counter (int at 0); [256..] h double
// buffer, 2 * BATCH*HIDDEN bf16 (256 KB)

__global__ void rnn_init(const float* __restrict__ h0, __bf16* __restrict__ hg0,
                         int* __restrict__ cnt) {
    int i = blockIdx.x * blockDim.x + threadIdx.x;
    if (i == 0) *cnt = 0;
    if (i < BATCH * HIDDEN) hg0[i] = (__bf16)h0[i & (HIDDEN - 1)];
}

__global__ void __launch_bounds__(THREADS, 1)
rnn_steps(const int* __restrict__ t, const float* __restrict__ emb,
          const float* __restrict__ W_ih, const float* __restrict__ W_hh,
          const float* __restrict__ W_proj, const float* __restrict__ b_proj,
          __bf16* __restrict__ hg, int* __restrict__ cnt,
          float* __restrict__ out) {
    extern __shared__ char smem_raw[];
    __bf16* Wl   = (__bf16*)smem_raw;            // [COLS_WG][KTOT]  80 KB
    __bf16* Abuf = Wl + COLS_WG * KTOT;          // [BATCH][KTOT]   160 KB

    const int wg   = blockIdx.x;
    const int tid  = threadIdx.x;
    const int wave = tid >> 5;
    const int lane = tid & 31;
    const int half = lane >> 4;   // 0: lanes 0-15, 1: lanes 16-31
    const int l16  = lane & 15;

    // ---- load this WG's weight slice once, f32 -> bf16, layout [n][k] ----
    for (int i = tid; i < COLS_WG * EMBED; i += THREADS) {
        int n = i >> 8, k = i & (EMBED - 1);
        Wl[n * KTOT + k] = (__bf16)W_ih[(wg * COLS_WG + n) * EMBED + k];
    }
    for (int i = tid; i < COLS_WG * HIDDEN; i += THREADS) {
        int n = i >> 10, k = i & (HIDDEN - 1);
        Wl[n * KTOT + EMBED + k] = (__bf16)W_hh[(wg * COLS_WG + n) * HIDDEN + k];
    }

    // one 16x16 output tile per wave: 4 M-tiles x 2 N-tiles = 8 waves
    const int mt = wave & 3;                 // M tile (batch rows)
    const int nt = wave >> 2;                // N tile within WG slice
    const int m0 = mt * 16;
    const int ncol = nt * 16 + l16;          // 0..31 within slice
    const __bf16* Brow = &Wl[ncol * KTOT];   // B frag: 16 contiguous k per lane

    for (int s = 0; s < SEQ; ++s) {
        const int cur = s & 1;
        const __bf16* hcur = hg + cur * (BATCH * HIDDEN);
        __bf16*       hnxt = hg + (cur ^ 1) * (BATCH * HIDDEN);

        // stage A = [x ; h] in LDS (bf16).
        // x: embedding gather, float4 load -> packed bf16x4 LDS store
        for (int i = tid; i < BATCH * (EMBED / 4); i += THREADS) {
            int b = i >> 6, q = i & 63;
            int ch = t[b * SEQ + s];
            const float4 f = ((const float4*)(emb + ch * EMBED))[q];
            bf16x4 v;
            v[0] = (__bf16)f.x; v[1] = (__bf16)f.y;
            v[2] = (__bf16)f.z; v[3] = (__bf16)f.w;
            *(bf16x4*)&Abuf[b * KTOT + q * 4] = v;
        }
        // h: previous-step bf16 state, async global -> LDS (no VGPR roundtrip)
#if __has_builtin(__builtin_amdgcn_global_load_async_to_lds_b128)
        for (int i = tid; i < BATCH * (HIDDEN / 8); i += THREADS) {
            int b = i >> 7, c = i & 127;
            __builtin_amdgcn_global_load_async_to_lds_b128(
                (ASG v4i*)(uintptr_t)(hcur + b * HIDDEN + c * 8),
                (ASL v4i*)(unsigned int)(uintptr_t)(Abuf + b * KTOT + EMBED + c * 8),
                0, 0);
        }
#if __has_builtin(__builtin_amdgcn_s_wait_asynccnt)
        __builtin_amdgcn_s_wait_asynccnt(0);
#else
        asm volatile("s_wait_asynccnt 0" ::: "memory");
#endif
#else
        for (int i = tid; i < BATCH * (HIDDEN / 8); i += THREADS) {
            int b = i >> 7, c = i & 127;
            ((uint4*)&Abuf[b * KTOT + EMBED])[c] =
                ((const uint4*)&hcur[b * HIDDEN])[c];
        }
#endif
        // prefetch next step's embedding rows behind this step's GEMM
        if (s + 1 < SEQ && tid < BATCH) {
            int chn = t[tid * SEQ + s + 1];
            __builtin_prefetch(emb + chn * EMBED, 0, 1);
        }
        __syncthreads();

        // 16x16 tile GEMM: 40 k-tiles of 32 (bf16 WMMA, f32 accumulate)
        f32x8 acc = {};
        const __bf16* Arow = &Abuf[(m0 + l16) * KTOT];
#pragma unroll 4
        for (int kt = 0; kt < KTOT / 32; ++kt) {
            // A fragment: k = half*16 + {0..15} contiguous (matches B)
            bf16x8 lo = *(const bf16x8*)(Arow + kt * 32 + half * 16);
            bf16x8 hi = *(const bf16x8*)(Arow + kt * 32 + half * 16 + 8);
            bf16x16 a;
#pragma unroll
            for (int j = 0; j < 8; ++j) { a[j] = lo[j]; a[j + 8] = hi[j]; }
            // B fragment: lane (n=l16, k = half*16 + {0..15}) contiguous
            bf16x16 bm = *(const bf16x16*)(Brow + kt * 32 + half * 16);
            acc = __builtin_amdgcn_wmma_f32_16x16x32_bf16(
                false, a, false, bm, (short)0, acc, false, false);
        }

        // tanh (native V_TANH_F32) + publish new h slice (bf16)
        const int jglob = wg * COLS_WG + ncol;
#pragma unroll
        for (int v = 0; v < 8; ++v) {
            int brow = m0 + v + 8 * half;          // C/D layout: m = v + 8*half
            hnxt[brow * HIDDEN + jglob] = (__bf16)fast_tanh(acc[v]);
        }

        // step sync: cluster barrier (NOP unless cluster-dispatched) + grid barrier
        __builtin_amdgcn_s_cluster_barrier();
        __threadfence();
        __syncthreads();
        if (tid == 0) {
            __hip_atomic_fetch_add(cnt, 1, __ATOMIC_ACQ_REL,
                                   __HIP_MEMORY_SCOPE_AGENT);
            const int target = NWG * (s + 1);
            while (__hip_atomic_load(cnt, __ATOMIC_ACQUIRE,
                                     __HIP_MEMORY_SCOPE_AGENT) < target)
                __builtin_amdgcn_s_sleep(1);
        }
        __syncthreads();
    }

    // ---- final projection: h(final) lives in buffer 0 after 512 steps ----
    // out[64x128] = h @ W_proj^T + b_proj.  WGs 0..7 x waves 0..3 cover it.
    if (wg < 8 && wave < 4) {
        const __bf16* hfin = hg;                      // buffer 0
        const int n = wg * 16 + l16;                  // output char column
        const float* wp = &W_proj[n * HIDDEN];
        const __bf16* Ar = &hfin[(wave * 16 + l16) * HIDDEN];
        f32x8 acc = {};
        for (int kt = 0; kt < HIDDEN / 32; ++kt) {
            bf16x8 lo = *(const bf16x8*)(Ar + kt * 32 + half * 16);
            bf16x8 hi = *(const bf16x8*)(Ar + kt * 32 + half * 16 + 8);
            bf16x16 a;
#pragma unroll
            for (int j = 0; j < 8; ++j) { a[j] = lo[j]; a[j + 8] = hi[j]; }
            bf16x16 bm;
            const float* wk = wp + kt * 32 + half * 16;
#pragma unroll
            for (int j = 0; j < 16; ++j) bm[j] = (__bf16)wk[j];
            acc = __builtin_amdgcn_wmma_f32_16x16x32_bf16(
                false, a, false, bm, (short)0, acc, false, false);
        }
        const float bias = b_proj[n];
#pragma unroll
        for (int v = 0; v < 8; ++v) {
            int brow = wave * 16 + v + 8 * half;
            out[brow * NCHAR + n] = acc[v] + bias;
        }
    }
}

extern "C" void kernel_launch(void* const* d_in, const int* in_sizes, int n_in,
                              void* d_out, int out_size, void* d_ws,
                              size_t ws_size, hipStream_t stream) {
    (void)in_sizes; (void)n_in; (void)out_size; (void)ws_size;
    const int*   t      = (const int*)d_in[0];
    const float* emb    = (const float*)d_in[1];
    const float* W_ih   = (const float*)d_in[2];
    const float* W_hh   = (const float*)d_in[3];
    const float* h0     = (const float*)d_in[4];
    const float* W_proj = (const float*)d_in[5];
    const float* b_proj = (const float*)d_in[6];
    float* out = (float*)d_out;

    int*    cnt = (int*)d_ws;
    __bf16* hg  = (__bf16*)((char*)d_ws + 256);

    const size_t lds_bytes =
        (size_t)(COLS_WG * KTOT + BATCH * KTOT) * sizeof(__bf16);  // 240 KB
    (void)hipFuncSetAttribute((const void*)rnn_steps,
                              hipFuncAttributeMaxDynamicSharedMemorySize,
                              (int)lds_bytes);

    rnn_init<<<(BATCH * HIDDEN + 255) / 256, 256, 0, stream>>>(h0, hg, cnt);
    rnn_steps<<<NWG, THREADS, lds_bytes, stream>>>(t, emb, W_ih, W_hh, W_proj,
                                                   b_proj, hg, cnt, out);
}
